// TinyLLM_24326694764997
// MI455X (gfx1250) — compile-verified, matrix operand order checked by reading, not run on hardware
//
#include <hip/hip_runtime.h>
#include <hip/hip_bf16.h>

#define SEQ 8192
#define DIM 1024

typedef __bf16 v16bf __attribute__((ext_vector_type(16)));
typedef float  v8f   __attribute__((ext_vector_type(8)));
typedef int    int4v __attribute__((ext_vector_type(4)));

typedef __attribute__((address_space(1))) int4v g_int4;
typedef __attribute__((address_space(3))) int4v l_int4;

__device__ __forceinline__ v8f wmma_bf16(v16bf a, v16bf b, v8f c) {
    // D = A(16x32 bf16) * B(32x16 bf16) + C(16x16 f32)
    return __builtin_amdgcn_wmma_f32_16x16x32_bf16(
        /*neg_a=*/false, a, /*neg_b=*/false, b,
        /*c_mod=*/(short)0, c, /*reuse_a=*/false, /*reuse_b=*/false);
}

__device__ __forceinline__ v8f v8f_zero() {
    v8f z;
    #pragma unroll
    for (int i = 0; i < 8; i++) z[i] = 0.0f;
    return z;
}

// ---------------------------------------------------------------------------
// Async global -> LDS 16-byte copy (ASYNCcnt-tracked, no VGPR round trip).
// ---------------------------------------------------------------------------
__device__ __forceinline__ void async_copy16(const void* g, void* l) {
#if __has_builtin(__builtin_amdgcn_global_load_async_to_lds_b128)
    __builtin_amdgcn_global_load_async_to_lds_b128(
        (g_int4*)g, (l_int4*)l, 0, 0);
#else
    unsigned loff = (unsigned)(unsigned long long)
        (__attribute__((address_space(3))) void*)l;
    unsigned long long ga = (unsigned long long)g;
    asm volatile("global_load_async_to_lds_b128 %0, %1, off"
                 :: "v"(loff), "v"(ga) : "memory");
#endif
}

template <int N>
__device__ __forceinline__ void wait_async() {
#if __has_builtin(__builtin_amdgcn_s_wait_asynccnt)
    __builtin_amdgcn_s_wait_asynccnt(N);
#else
    asm volatile("s_wait_asynccnt %0" :: "n"(N) : "memory");
#endif
}

// ---------------------------------------------------------------------------
// f32 -> bf16 conversion (grid-strided)
// ---------------------------------------------------------------------------
__global__ void f32_to_bf16_kernel(const float* __restrict__ src,
                                   __bf16* __restrict__ dst, long n) {
    long i = (long)blockIdx.x * blockDim.x + threadIdx.x;
    long stride = (long)gridDim.x * blockDim.x;
    for (; i < n; i += stride) dst[i] = (__bf16)src[i];
}

// ---------------------------------------------------------------------------
// C[m,n] = sum_k A[m,k] * B[n,k]   (i.e. A @ B^T, B row-major [N][K])
// WG: 256 threads (8 waves), tile 128(M) x 64(N).
// K staged in 64-chunks, double-buffered via async-to-LDS (6 ops/thread/chunk).
// Each wave: 16 rows x 64 cols = 4 16x16 f32 accumulators.
// ---------------------------------------------------------------------------
template <typename OutT>
__global__ __launch_bounds__(256) void gemm_bt(const __bf16* __restrict__ A,
                                               const __bf16* __restrict__ B,
                                               OutT* __restrict__ C,
                                               int M, int N, int K) {
    __shared__ __align__(16) __bf16 As[2][128 * 64];
    __shared__ __align__(16) __bf16 Bs[2][64 * 64];

    const int tid  = threadIdx.x;
    const int lane = tid & 31;
    const int w    = tid >> 5;
    const int hi   = lane >> 4;   // K-half of the 32-chunk this lane holds
    const int lr   = lane & 15;   // row (A) / col (B) selector within tile

    const long mbase = (long)blockIdx.y * 128;
    const long nbase = (long)blockIdx.x * 64;

    v8f acc[4];
    #pragma unroll
    for (int i = 0; i < 4; i++) acc[i] = v8f_zero();

    const int nC = K / 64;

    // Issue chunk 0 (A: 4 ops/thread, B: 2 ops/thread)
    for (int idx = tid; idx < 1024; idx += 256) {
        int r = idx >> 3, g = idx & 7;
        async_copy16(&A[(mbase + r) * K + g * 8], &As[0][r * 64 + g * 8]);
    }
    for (int idx = tid; idx < 512; idx += 256) {
        int r = idx >> 3, g = idx & 7;
        async_copy16(&B[(nbase + r) * K + g * 8], &Bs[0][r * 64 + g * 8]);
    }

    for (int c = 0; c < nC; c++) {
        const int cur = c & 1, nxt = cur ^ 1;
        if (c + 1 < nC) {
            int kc = (c + 1) * 64;
            for (int idx = tid; idx < 1024; idx += 256) {
                int r = idx >> 3, g = idx & 7;
                async_copy16(&A[(mbase + r) * K + kc + g * 8],
                             &As[nxt][r * 64 + g * 8]);
            }
            for (int idx = tid; idx < 512; idx += 256) {
                int r = idx >> 3, g = idx & 7;
                async_copy16(&B[(nbase + r) * K + kc + g * 8],
                             &Bs[nxt][r * 64 + g * 8]);
            }
            wait_async<6>();   // <=6 outstanding => chunk c landed
        } else {
            wait_async<0>();
        }
        __syncthreads();

        #pragma unroll
        for (int kk = 0; kk < 64; kk += 32) {
            v16bf a = *(const v16bf*)&As[cur][(w * 16 + lr) * 64 + kk + hi * 16];
            #pragma unroll
            for (int nt = 0; nt < 4; nt++) {
                v16bf b = *(const v16bf*)&Bs[cur][(nt * 16 + lr) * 64 + kk + hi * 16];
                acc[nt] = wmma_bf16(a, b, acc[nt]);
            }
        }
        __syncthreads();   // all waves done reading before buffer is rewritten
    }

    // Epilogue: C/D layout -> VGPR r holds M = r + 8*hi, lane lr holds N
    #pragma unroll
    for (int nt = 0; nt < 4; nt++) {
        #pragma unroll
        for (int r = 0; r < 8; r++) {
            long row = mbase + w * 16 + r + hi * 8;
            long col = nbase + nt * 16 + lr;
            C[row * N + col] = (OutT)acc[nt][r];
        }
    }
}

// ---------------------------------------------------------------------------
// Flash attention, causal, single head, d=1024.
// One WG per Br=64 query rows. 8 waves (256 threads).
// LDS: Qs 64x1024 bf16 | Vt 1024x64 bf16 (transposed) | Ks 2 x 64x128 bf16 |
//      Ss 64x64 f32 | Ps 64x64 bf16 | m/l/alpha 64 f32 each.  Total 320256 B.
// Scores: K streamed in double-buffered 128-wide async chunks.
// PV: each wave owns 128 of 1024 O columns (32 v8f accumulators = 256 VGPRs).
// ---------------------------------------------------------------------------
#define FLASH_LDS_BYTES 320256

__global__ __launch_bounds__(256) void flash_kernel(const __bf16* __restrict__ Qg,
                                                    const __bf16* __restrict__ Kg,
                                                    const __bf16* __restrict__ Vg,
                                                    __bf16* __restrict__ Og) {
    extern __shared__ char smem[];
    __bf16* Qs  = (__bf16*)(smem);              // 131072 B
    __bf16* Vt  = (__bf16*)(smem + 131072);     // 131072 B
    __bf16* Ks0 = (__bf16*)(smem + 262144);     //  16384 B
    __bf16* Ks1 = (__bf16*)(smem + 278528);     //  16384 B
    float*  Ss  = (float*) (smem + 294912);     //  16384 B
    __bf16* Ps  = (__bf16*)(smem + 311296);     //   8192 B
    float*  m_s = (float*) (smem + 319488);     //    256 B
    float*  l_s = (float*) (smem + 319744);     //    256 B
    float*  a_s = (float*) (smem + 320000);     //    256 B
    __bf16* Ks[2] = { Ks0, Ks1 };

    const int tid  = threadIdx.x;
    const int lane = tid & 31;
    const int w    = tid >> 5;
    const int hi   = lane >> 4;
    const int lr   = lane & 15;
    const int qblk = blockIdx.x;
    const long qbase = (long)qblk * 64;
    const int mt = w >> 1;           // score-phase row tile of this wave

    // Stage the whole Q tile (64 x 1024 bf16) asynchronously, once.
    for (int idx = tid; idx < 8192; idx += 256) {
        int r = idx >> 7, g = idx & 127;
        async_copy16(&Qg[(qbase + r) * 1024 + g * 8], &Qs[r * 1024 + g * 8]);
    }
    if (tid < 64) { m_s[tid] = -1e30f; l_s[tid] = 0.0f; }

    v8f oacc[4][8];
    #pragma unroll
    for (int rt = 0; rt < 4; rt++)
        #pragma unroll
        for (int ct = 0; ct < 8; ct++) oacc[rt][ct] = v8f_zero();

    wait_async<0>();
    __syncthreads();

    for (int j = 0; j <= qblk; j++) {
        // ---- Phase 1: S = Q K^T, K streamed in 8 chunks of 128, 2-deep ---
        v8f sacc[2];
        sacc[0] = v8f_zero(); sacc[1] = v8f_zero();

        // Prologue: chunk 0 (4 async ops/thread)
        for (int idx = tid; idx < 1024; idx += 256) {
            int r = idx >> 4, g = idx & 15;
            async_copy16(&Kg[((long)j * 64 + r) * 1024 + g * 8],
                         &Ks[0][r * 128 + g * 8]);
        }
        for (int c = 0; c < 8; c++) {
            const int cur = c & 1, nxt = cur ^ 1;
            if (c + 1 < 8) {
                int dc = (c + 1) * 128;
                for (int idx = tid; idx < 1024; idx += 256) {
                    int r = idx >> 4, g = idx & 15;
                    async_copy16(&Kg[((long)j * 64 + r) * 1024 + dc + g * 8],
                                 &Ks[nxt][r * 128 + g * 8]);
                }
                wait_async<4>();   // chunk c landed, chunk c+1 in flight
            } else {
                wait_async<0>();
            }
            __syncthreads();

            const int dc = c * 128;
            #pragma unroll
            for (int kk = 0; kk < 128; kk += 32) {
                v16bf a = *(const v16bf*)&Qs[(mt * 16 + lr) * 1024 + dc + kk + hi * 16];
                #pragma unroll
                for (int t = 0; t < 2; t++) {
                    int nt = (w & 1) * 2 + t;
                    v16bf b = *(const v16bf*)&Ks[cur][(nt * 16 + lr) * 128 + kk + hi * 16];
                    sacc[t] = wmma_bf16(a, b, sacc[t]);
                }
            }
            __syncthreads();   // done reading cur before it is rewritten
        }

        // ---- Scale + causal mask, spill S tile to LDS -------------------
        #pragma unroll
        for (int t = 0; t < 2; t++) {
            int nt = (w & 1) * 2 + t;
            #pragma unroll
            for (int r = 0; r < 8; r++) {
                int lrow = mt * 16 + r + hi * 8;
                int lcol = nt * 16 + lr;
                float v = sacc[t][r] * 0.03125f;   // 1/sqrt(1024)
                long gr = qbase + lrow;
                long gc = (long)j * 64 + lcol;
                if (gc > gr) v = -1e30f;
                Ss[lrow * 64 + lcol] = v;
            }
        }
        __syncthreads();

        // ---- Online softmax row update (one thread per row) -------------
        if (tid < 64) {
            float mold = m_s[tid], mnew = mold;
            for (int c = 0; c < 64; c++) mnew = fmaxf(mnew, Ss[tid * 64 + c]);
            float alpha = __expf(mold - mnew);
            float sum = 0.0f;
            for (int c = 0; c < 64; c++) {
                float p = __expf(Ss[tid * 64 + c] - mnew);
                Ps[tid * 64 + c] = (__bf16)p;
                sum += p;
            }
            m_s[tid] = mnew;
            l_s[tid] = l_s[tid] * alpha + sum;
            a_s[tid] = alpha;
        }

        // ---- Stage V transposed: Vt[d][k] so PV B-frags are contiguous --
        for (int idx = tid; idx < 8192; idx += 256) {
            int k = idx >> 7, g = idx & 127;
            __bf16 tmp[8];
            *(int4*)tmp = *(const int4*)&Vg[((long)j * 64 + k) * 1024 + g * 8];
            #pragma unroll
            for (int e = 0; e < 8; e++) Vt[(g * 8 + e) * 64 + k] = tmp[e];
        }
        __syncthreads();

        // ---- Rescale O by exp(m_old - m_new) ----------------------------
        #pragma unroll
        for (int rt = 0; rt < 4; rt++) {
            #pragma unroll
            for (int r = 0; r < 8; r++) {
                float alpha = a_s[rt * 16 + r + hi * 8];
                #pragma unroll
                for (int ct = 0; ct < 8; ct++) oacc[rt][ct][r] *= alpha;
            }
        }

        // ---- Phase 2: O += P V over this wave's 128 columns -------------
        #pragma unroll
        for (int kk = 0; kk < 64; kk += 32) {
            v16bf bfrag[8];
            #pragma unroll
            for (int ct = 0; ct < 8; ct++)
                bfrag[ct] = *(const v16bf*)&Vt[(w * 128 + ct * 16 + lr) * 64 + kk + hi * 16];
            #pragma unroll
            for (int rt = 0; rt < 4; rt++) {
                v16bf a = *(const v16bf*)&Ps[(rt * 16 + lr) * 64 + kk + hi * 16];
                #pragma unroll
                for (int ct = 0; ct < 8; ct++)
                    oacc[rt][ct] = wmma_bf16(a, bfrag[ct], oacc[rt][ct]);
            }
        }
        __syncthreads();
    }

    // ---- Normalize by row sum l and emit bf16 O for the Wo projection ----
    #pragma unroll
    for (int rt = 0; rt < 4; rt++) {
        #pragma unroll
        for (int r = 0; r < 8; r++) {
            int lrow = rt * 16 + r + hi * 8;
            float inv = 1.0f / l_s[lrow];
            long grow = qbase + lrow;
            #pragma unroll
            for (int ct = 0; ct < 8; ct++) {
                long gcol = w * 128 + ct * 16 + lr;
                Og[grow * 1024 + gcol] = (__bf16)(oacc[rt][ct][r] * inv);
            }
        }
    }
}

// ---------------------------------------------------------------------------
// Launch
// ---------------------------------------------------------------------------
extern "C" void kernel_launch(void* const* d_in, const int* in_sizes, int n_in,
                              void* d_out, int out_size, void* d_ws, size_t ws_size,
                              hipStream_t stream) {
    const float* x  = (const float*)d_in[0];
    const float* wq = (const float*)d_in[1];
    const float* wk = (const float*)d_in[2];
    const float* wv = (const float*)d_in[3];
    const float* wo = (const float*)d_in[4];
    float* out = (float*)d_out;

    const long SD = (long)SEQ * DIM;   // 8,388,608
    const long DD = (long)DIM * DIM;   // 1,048,576

    // Workspace layout (bf16): Xb | Wq | Wk | Wv | Wo | Q | K | V | O  (~88 MB)
    __bf16* Xb  = (__bf16*)d_ws;
    __bf16* Wqb = Xb  + SD;
    __bf16* Wkb = Wqb + DD;
    __bf16* Wvb = Wkb + DD;
    __bf16* Wob = Wvb + DD;
    __bf16* Qb  = Wob + DD;
    __bf16* Kb  = Qb  + SD;
    __bf16* Vb  = Kb  + SD;
    __bf16* Ob  = Vb  + SD;

    f32_to_bf16_kernel<<<1024, 256, 0, stream>>>(x,  Xb,  SD);
    f32_to_bf16_kernel<<<256,  256, 0, stream>>>(wq, Wqb, DD);
    f32_to_bf16_kernel<<<256,  256, 0, stream>>>(wk, Wkb, DD);
    f32_to_bf16_kernel<<<256,  256, 0, stream>>>(wv, Wvb, DD);
    f32_to_bf16_kernel<<<256,  256, 0, stream>>>(wo, Wob, DD);

    dim3 grid(DIM / 64, SEQ / 128);
    gemm_bt<__bf16><<<grid, 256, 0, stream>>>(Xb, Wqb, Qb, SEQ, DIM, DIM);
    gemm_bt<__bf16><<<grid, 256, 0, stream>>>(Xb, Wkb, Kb, SEQ, DIM, DIM);
    gemm_bt<__bf16><<<grid, 256, 0, stream>>>(Xb, Wvb, Vb, SEQ, DIM, DIM);

    (void)hipFuncSetAttribute((const void*)flash_kernel,
                              hipFuncAttributeMaxDynamicSharedMemorySize,
                              FLASH_LDS_BYTES);
    flash_kernel<<<SEQ / 64, 256, FLASH_LDS_BYTES, stream>>>(Qb, Kb, Vb, Ob);

    gemm_bt<float><<<grid, 256, 0, stream>>>(Ob, Wob, out, SEQ, DIM, DIM);
}
